// Model_45913200394534
// MI455X (gfx1250) — matrix operand order repaired and checked by reading, not compiled, not run on hardware
//
#include <hip/hip_runtime.h>
#include <hip/hip_bf16.h>
#include <math.h>

typedef __attribute__((ext_vector_type(16))) _Float16 v16h;
typedef __attribute__((ext_vector_type(8)))  _Float16 v8h;
typedef __attribute__((ext_vector_type(8)))  float    v8f;

#define BB 64
#define KK 32
#define AA 50
#define DD 100
#define HH 100
#define KP 224   // padded K-dim (2*D=200 -> 7*32)
#define NP 112   // padded N-dim (H=100 -> 7*16)
#define MP 64    // padded M-dim (A=50 -> 4*16)
#define NBR_PITCH 113

// workspace byte offsets (256-aligned)
#define WS_HALL    ((size_t)0)                       // float[6*B*K*H]  = 4,915,200 B
#define WS_SUPMEAN ((size_t)4915200)                 // float[B*2H]     = 51,200 B
#define WS_WNT     ((size_t)4966400)                 // half[NP*KP]     = 50,176 B

#define LOG2E      1.4426950408889634f
#define TWO_LOG2E  2.8853900817779268f

// branch-free tanh: tanh(x) = sign(x) * (1 - 2/(1 + exp(2|x|)))
// lowers to v_exp_f32 + v_rcp_f32 + fma + v_bfi_b32; saturates naturally (exp2->inf, rcp->0)
__device__ __forceinline__ float fast_tanh(float x) {
    float ax = __builtin_fabsf(x);
    float e  = __builtin_amdgcn_exp2f(ax * TWO_LOG2E);          // exp(2|x|)
    float r  = __builtin_fmaf(-2.f, __builtin_amdgcn_rcpf(1.f + e), 1.f);
    return __builtin_copysignf(r, x);
}

__device__ __forceinline__ float fast_exp(float x) {
    return __builtin_amdgcn_exp2f(x * LOG2E);
}

// ---------------- kernel 1: Wn -> WnT (f16, transposed, padded) ----------------
__global__ void prep_wnt_kernel(const float* __restrict__ Wn, _Float16* __restrict__ WnT) {
    int idx = blockIdx.x * blockDim.x + threadIdx.x;
    if (idx >= NP * KP) return;
    int n  = idx / KP;
    int kk = idx % KP;
    float v = 0.f;
    if (n < HH && kk < 2 * DD) v = Wn[kk * HH + n];   // Wn is (200,100): Wn[kk][n]
    WnT[idx] = (_Float16)v;
}

// ---------------- kernel 2: fused gather + WMMA GEMM + attention encode ----------------
__global__ __launch_bounds__(128) void encode_kernel(
    const int* __restrict__ sup, const int* __restrict__ pos, const int* __restrict__ neg,
    const int* __restrict__ m0, const int* __restrict__ m1, const int* __restrict__ m2,
    const int* __restrict__ m3, const int* __restrict__ m4, const int* __restrict__ m5,
    const float* __restrict__ ent, const float* __restrict__ rel,
    const float* __restrict__ Ws, const _Float16* __restrict__ WnT,
    float* __restrict__ h_all)
{
    __shared__ _Float16 Ash[MP * KP];          // 28672 B: gathered [rel||adj] tile, f16
    __shared__ float    nbr[MP * NBR_PITCH];   // 28928 B: tanh(GEMM) result
    __shared__ float    qsh[NP];
    __shared__ float    esh[NP];
    __shared__ float    ssh[MP];
    __shared__ float    wsh[MP];
    __shared__ float    msh[MP];
    __shared__ int      relidx[MP];
    __shared__ int      adjidx[MP];

    const int tid  = threadIdx.x;
    const int blk  = blockIdx.x;
    const int call = blk / (BB * KK);
    const int rem  = blk % (BB * KK);
    const int b    = rem / KK;
    const int k    = rem % KK;
    const int side = call & 1;

    const int* pair = (call < 2) ? sup : (call < 4) ? pos : neg;
    const int* meta;
    switch (call) {
        case 0: meta = m0; break; case 1: meta = m1; break;
        case 2: meta = m2; break; case 3: meta = m3; break;
        case 4: meta = m4; break; default: meta = m5; break;
    }
    const int  id   = pair[(b * 2 + side) * KK + k];
    const int* mrel = meta + (((size_t)b * 2 + 0) * KK + k) * AA;
    const int* madj = meta + (((size_t)b * 2 + 1) * KK + k) * AA;

    // phase 0: indices, mask, entity row
    if (tid < MP) {
        int ri = 0, ai = 0;
        if (tid < AA) { ri = mrel[tid]; ai = madj[tid]; }
        relidx[tid] = ri;
        adjidx[tid] = ai;
        msh[tid] = (tid < AA && ri != 0) ? 1.f : 0.f;
    }
    if (tid < DD) esh[tid] = ent[(size_t)id * DD + tid];
    __syncthreads();

    // phase 1: gather [rel || adj] -> f16 LDS tile (zero padded)
    for (int idx = tid; idx < MP * KP; idx += 128) {
        int a = idx / KP, j = idx % KP;
        float v = 0.f;
        if (a < AA && j < 2 * DD) {
            v = (j < DD) ? rel[(size_t)relidx[a] * DD + j]
                         : ent[(size_t)adjidx[a] * DD + (j - DD)];
        }
        Ash[idx] = (_Float16)v;
    }
    // phase 1b: q = e @ Ws  (1x100 @ 100x100, VALU)
    if (tid < HH) {
        float q = 0.f;
        for (int d = 0; d < DD; ++d) q += esh[d] * Ws[d * HH + tid];
        qsh[tid] = q;
    }
    __syncthreads();

    // phase 2: WMMA GEMM  nbr_raw = A(64x224,f16) @ WnT^T(224x112,f16), f32 acc
    {
        const int wave = tid >> 5;
        const int lane = tid & 31;
        const int lrow = lane & 15;
        const int lhi  = (lane >> 4) & 1;
        const int mt   = wave;                        // each wave owns one 16-row M tile
        const _Float16* arow = Ash + (mt * 16 + lrow) * KP;
        for (int nt = 0; nt < 7; ++nt) {
            v8f acc = {0.f, 0.f, 0.f, 0.f, 0.f, 0.f, 0.f, 0.f};
            const _Float16* brow = WnT + (size_t)(nt * 16 + lrow) * KP;
            for (int kt = 0; kt < 7; ++kt) {
                const int k0 = kt * 32;
                // A frag: lanes 0-15 -> K[k0..k0+7],[k0+16..k0+23]; lanes 16-31 -> +8
                v8h alo = *(const v8h*)(arow + k0 + lhi * 8);
                v8h ahi = *(const v8h*)(arow + k0 + 16 + lhi * 8);
                v16h afrag = __builtin_shufflevector(alo, ahi,
                    0,1,2,3,4,5,6,7,8,9,10,11,12,13,14,15);
                // B frag: lanes 0-15 -> K[k0..k0+7],[k0+16..k0+23]; lanes 16-31 -> +8
                v8h blo = *(const v8h*)(brow + k0 + lhi * 8);
                v8h bhi = *(const v8h*)(brow + k0 + 16 + lhi * 8);
                v16h bfrag = __builtin_shufflevector(blo, bhi,
                    0,1,2,3,4,5,6,7,8,9,10,11,12,13,14,15);
                acc = __builtin_amdgcn_wmma_f32_16x16x32_f16(
                    false, afrag, false, bfrag, (short)0, acc, false, false);
            }
            // C layout: lane<16 -> M=r, lane>=16 -> M=8+r ; N = lane%16
            const int n = nt * 16 + lrow;
#pragma unroll
            for (int r = 0; r < 8; ++r) {
                const int m = mt * 16 + lhi * 8 + r;
                nbr[m * NBR_PITCH + n] = fast_tanh(acc[r]);
            }
        }
    }
    __syncthreads();

    // phase 3: attention scores + masked softmax
    if (tid < AA) {
        float s = 0.f;
        for (int h = 0; h < HH; ++h) s += nbr[tid * NBR_PITCH + h] * qsh[h];
        s *= 0.1f;                                    // 1/sqrt(H)
        ssh[tid] = (msh[tid] != 0.f) ? s : -1.0e9f;
    }
    __syncthreads();
    if (tid == 0) {
        float mx = -INFINITY;
        for (int a = 0; a < AA; ++a) mx = fmaxf(mx, ssh[a]);
        float sum = 0.f;
        for (int a = 0; a < AA; ++a) { float e2 = fast_exp(ssh[a] - mx); wsh[a] = e2; sum += e2; }
        float inv = __builtin_amdgcn_rcpf(sum);
        for (int a = 0; a < AA; ++a) wsh[a] = wsh[a] * inv * msh[a];
    }
    __syncthreads();

    // phase 4: agg + output tanh
    if (tid < HH) {
        float agg = 0.f;
        for (int a = 0; a < AA; ++a) agg += wsh[a] * nbr[a * NBR_PITCH + tid];
        float o = fast_tanh(agg + qsh[tid]);
        h_all[(((size_t)call * BB + b) * KK + k) * HH + tid] = o;
    }
}

// ---------------- kernel 3: mean of support over K ----------------
__global__ void supmean_kernel(const float* __restrict__ h_all, float* __restrict__ sup_mean) {
    int b = blockIdx.x;
    int j = threadIdx.x;
    if (j >= 2 * HH) return;
    int c  = (j < HH) ? 0 : 1;
    int jj = (j < HH) ? j : j - HH;
    float s = 0.f;
    for (int k = 0; k < KK; ++k)
        s += h_all[(((size_t)c * BB + b) * KK + k) * HH + jj];
    sup_mean[b * 2 * HH + j] = s * (1.f / (float)KK);
}

// ---------------- kernel 4: comparator MLP ----------------
__global__ __launch_bounds__(256) void compare_kernel(
    const float* __restrict__ h_all, const float* __restrict__ sup_mean,
    const float* __restrict__ Wc1, const float* __restrict__ bc1,
    const float* __restrict__ Wc2, float* __restrict__ out)
{
    __shared__ float xsh[4 * HH];
    __shared__ float red[256];
    const int blk = blockIdx.x;
    const int s   = blk / (BB * KK);      // 0 = pos, 1 = neg
    const int rem = blk % (BB * KK);
    const int b   = rem / KK;
    const int k   = rem % KK;
    const int tid = threadIdx.x;

    for (int t = tid; t < 4 * HH; t += 256) {
        if (t < 2 * HH) {
            xsh[t] = sup_mean[b * 2 * HH + t];
        } else {
            int j = t - 2 * HH;
            int c = (s ? 4 : 2) + ((j < HH) ? 0 : 1);
            xsh[t] = h_all[(((size_t)c * BB + b) * KK + k) * HH + (j % HH)];
        }
    }
    __syncthreads();

    float contrib = 0.f;
    if (tid < 2 * HH) {
        float hv = bc1[tid];
        for (int i = 0; i < 4 * HH; ++i) hv += xsh[i] * Wc1[i * (2 * HH) + tid];
        hv = fmaxf(hv, 0.f);
        contrib = hv * Wc2[tid];
    }
    red[tid] = contrib;
    __syncthreads();
    for (int off = 128; off > 0; off >>= 1) {
        if (tid < off) red[tid] += red[tid + off];
        __syncthreads();
    }
    if (tid == 0) out[(size_t)s * BB * KK + b * KK + k] = red[0];
}

// ---------------- launch ----------------
extern "C" void kernel_launch(void* const* d_in, const int* in_sizes, int n_in,
                              void* d_out, int out_size, void* d_ws, size_t ws_size,
                              hipStream_t stream) {
    const int*   sup = (const int*)d_in[0];
    const int*   pos = (const int*)d_in[1];
    const int*   neg = (const int*)d_in[2];
    const int*   m0  = (const int*)d_in[3];
    const int*   m1  = (const int*)d_in[4];
    const int*   m2  = (const int*)d_in[5];
    const int*   m3  = (const int*)d_in[6];
    const int*   m4  = (const int*)d_in[7];
    const int*   m5  = (const int*)d_in[8];
    const float* ent = (const float*)d_in[9];
    const float* rel = (const float*)d_in[10];
    const float* Wn  = (const float*)d_in[11];
    const float* Ws  = (const float*)d_in[12];
    const float* Wc1 = (const float*)d_in[13];
    const float* bc1 = (const float*)d_in[14];
    const float* Wc2 = (const float*)d_in[15];
    float* out = (float*)d_out;

    char* ws = (char*)d_ws;
    float*    h_all    = (float*)(ws + WS_HALL);
    float*    sup_mean = (float*)(ws + WS_SUPMEAN);
    _Float16* WnT      = (_Float16*)(ws + WS_WNT);

    (void)in_sizes; (void)n_in; (void)out_size; (void)ws_size;

    prep_wnt_kernel<<<(NP * KP + 255) / 256, 256, 0, stream>>>(Wn, WnT);
    encode_kernel<<<6 * BB * KK, 128, 0, stream>>>(
        sup, pos, neg, m0, m1, m2, m3, m4, m5, ent, rel, Ws, WnT, h_all);
    supmean_kernel<<<BB, 256, 0, stream>>>(h_all, sup_mean);
    compare_kernel<<<2 * BB * KK, 256, 0, stream>>>(h_all, sup_mean, Wc1, bc1, Wc2, out);
}
